// LocalWindowAttention3D_1829656068314
// MI455X (gfx1250) — matrix operand order repaired and checked by reading, not compiled, not run on hardware
//
#include <hip/hip_runtime.h>

// ---------------------------------------------------------------------------
// 3D local window attention, MI455X (gfx1250, wave32, WMMA bf16).
// Pipeline: [cvt weights -> bf16] -> [QKV gemm, windowed layout] ->
//           [per-window attention] -> [output projection + bias].
// All matmuls use v_wmma_f32_16x16x32_bf16 (f32 accumulate).
// Workspace usage: 131072 + 4*56623104 = 226,623,488 bytes.
// ---------------------------------------------------------------------------

#define DIMC   128
#define HEADS  8
#define HD     16
#define NWAX   12          // windows per axis (48/4)
#define NWIN   1728        // 12^3
#define TOK    64          // 4^3 tokens per window
#define DD     48
#define SPT    110592      // 48^3 positions per batch
#define NB     2
#define SCALE  0.08838834764831845f   // 128^-0.5 (reference scales by dim, not head_dim)

typedef __bf16 bf16x16 __attribute__((ext_vector_type(16)));
typedef float  f32x8   __attribute__((ext_vector_type(8)));
typedef float  f32x4   __attribute__((ext_vector_type(4)));
typedef int    i32x4   __attribute__((ext_vector_type(4)));
typedef unsigned short u16t;

union Frag {
    bf16x16 v;
    i32x4   q[2];
    u16t    h[16];
};

__device__ __forceinline__ u16t f2bf(float f) {
    unsigned int x = __float_as_uint(f);
    unsigned int r = x + 0x7FFFu + ((x >> 16) & 1u);   // round-to-nearest-even
    return (u16t)(r >> 16);
}

__device__ __forceinline__ f32x8 wmma_bf16(const Frag& a, const Frag& b, f32x8 c) {
    return __builtin_amdgcn_wmma_f32_16x16x32_bf16(false, a.v, false, b.v,
                                                   (short)0, c, false, false);
}

// ---------------------------------------------------------------------------
// Kernel 0: convert Wqkv (384x128) and Wout (128x128) to bf16 (L2-resident).
// ---------------------------------------------------------------------------
__global__ __launch_bounds__(256) void cvt_weights(
        const float* __restrict__ Wqkv, const float* __restrict__ Wout,
        u16t* __restrict__ wq, u16t* __restrict__ wo) {
    int i = blockIdx.x * 256 + threadIdx.x;
    if (i < 3 * DIMC * DIMC) wq[i] = f2bf(Wqkv[i]);
    if (i < DIMC * DIMC)     wo[i] = f2bf(Wout[i]);
}

// ---------------------------------------------------------------------------
// Kernel 1: QKV projection. One block per (batch, window).
// Output layouts (unit = (b*8+h)*NWIN+win):
//   Q[unit][tok][j], K[unit][tok][j], Vt[unit][j][tok]  (all bf16)
// Each wave has a fixed M-tile (m0 = (wave&3)*16) -> A fragments loaded once;
// B fragments for each N-tile are batch-loaded, then 4 WMMAs issue together.
// ---------------------------------------------------------------------------
#define XPAD 136   // 64 rows * 136 bf16; 272B row stride -> conflict-free frag reads
__global__ __launch_bounds__(256) void qkv_kernel(
        const float* __restrict__ x, const u16t* __restrict__ wq,
        u16t* __restrict__ Q, u16t* __restrict__ K, u16t* __restrict__ Vt) {
    __shared__ u16t xt[TOK * XPAD];
    const int tid = threadIdx.x;
    const int blk = blockIdx.x;
    const int b   = blk / NWIN;
    const int win = blk % NWIN;
    const int d0 = (win / 144) * 4;
    const int h0 = ((win / NWAX) % NWAX) * 4;
    const int w0 = (win % NWAX) * 4;

    // Stage x tile (64 tokens x 128 channels), f32 -> bf16.
    // Each thread loads one float4 (4 contiguous w positions, 16B aligned).
    for (int it = 0; it < 8; ++it) {
        int idx = it * 256 + tid;                  // 2048 float4 chunks
        int tr = idx & 15, c = idx >> 4;           // tr = token row (wd,wh)
        int wd = tr >> 2, wh = tr & 3;
        int pos = (d0 + wd) * (DD * DD) + (h0 + wh) * DD + w0;
        f32x4 v = *(const f32x4*)&x[(size_t)(b * DIMC + c) * SPT + pos];
        u16t* dst = &xt[(tr * 4) * XPAD + c];
        dst[0 * XPAD] = f2bf(v[0]);
        dst[1 * XPAD] = f2bf(v[1]);
        dst[2 * XPAD] = f2bf(v[2]);
        dst[3 * XPAD] = f2bf(v[3]);
    }
    __syncthreads();

    const int wave = tid >> 5, lane = tid & 31;
    const int lh = lane >> 4, lm = lane & 15;
    const int m0 = (wave & 3) * 16;                // fixed M-tile per wave

    // A fragments for all 4 K-steps (loaded once, reused by 12 N-tiles)
    Frag a[4];
    #pragma unroll
    for (int k = 0; k < 4; ++k) {
        a[k].q[0] = *(const i32x4*)&xt[(m0 + lm) * XPAD + k * 32 + 16 * lh];
        a[k].q[1] = *(const i32x4*)&xt[(m0 + lm) * XPAD + k * 32 + 16 * lh + 8];
    }

    for (int i = 0; i < 12; ++i) {
        const int id = wave + 8 * i;               // 0..95
        const int n0 = (id >> 2) * 16;             // N-tile 0..23
        // Batch-load all 4 B fragments, then issue 4 WMMAs back-to-back.
        Frag bm[4];
        #pragma unroll
        for (int k = 0; k < 4; ++k) {
            const i32x4* bp = (const i32x4*)&wq[(n0 + lm) * DIMC + k * 32 + 16 * lh];
            bm[k].q[0] = bp[0]; bm[k].q[1] = bp[1];
        }
        f32x8 acc = {0.f, 0.f, 0.f, 0.f, 0.f, 0.f, 0.f, 0.f};
        #pragma unroll
        for (int k = 0; k < 4; ++k) acc = wmma_bf16(a[k], bm[k], acc);

        // Scatter: e in [n0, n0+16) never crosses a qkv/head boundary.
        const int e   = n0 + lm;
        const int sel = e >> 7;                    // 0=q 1=k 2=v
        const int hh  = (e >> 4) & 7;
        const int j   = e & 15;
        const size_t ub = (size_t)(b * HEADS + hh) * NWIN + win;
        if (sel == 0) {
            u16t* dst = Q + ub * (TOK * HD);
            #pragma unroll
            for (int r = 0; r < 8; ++r) dst[(m0 + r + 8 * lh) * HD + j] = f2bf(acc[r]);
        } else if (sel == 1) {
            u16t* dst = K + ub * (TOK * HD);
            #pragma unroll
            for (int r = 0; r < 8; ++r) dst[(m0 + r + 8 * lh) * HD + j] = f2bf(acc[r]);
        } else {
            u16t* dst = Vt + ub * (HD * TOK);      // transposed: [j][token]
            #pragma unroll
            for (int r = 0; r < 8; ++r) dst[j * TOK + (m0 + r + 8 * lh)] = f2bf(acc[r]);
        }
    }
}

// ---------------------------------------------------------------------------
// Kernel 2: attention per (b, head, window). 4 waves, each owns 16 query rows.
// S = Q K^T (head_dim 16 zero-padded to K=32), f32 softmax, O = P V.
// Ao[b][win*64+t][c] bf16 (token order == window order, per reference).
// ---------------------------------------------------------------------------
#define PROW 72    // P tile row stride (144B) -> conflict-free ds reads
__global__ __launch_bounds__(128) void attn_kernel(
        const u16t* __restrict__ Q, const u16t* __restrict__ K,
        const u16t* __restrict__ Vt, u16t* __restrict__ Ao) {
    __shared__ u16t pt[4 * 16 * PROW];
    const int tid = threadIdx.x;
    const int wave = tid >> 5, lane = tid & 31;
    const int lh = lane >> 4, lm = lane & 15;
    const int unit = blockIdx.x;
    const int b   = unit / (HEADS * NWIN);
    const int rem = unit % (HEADS * NWIN);
    const int h   = rem / NWIN;
    const int win = rem % NWIN;
    const int m0  = wave * 16;
    const size_t base = (size_t)unit * (TOK * HD);
    const i32x4 zero = {0, 0, 0, 0};

    // Q A-fragment: lanes<16 take j=0..15, lanes>=16 (j=16..31) are zero padding.
    Frag qa;
    qa.q[0] = lh ? zero : *(const i32x4*)&Q[base + (m0 + lm) * HD];
    qa.q[1] = lh ? zero : *(const i32x4*)&Q[base + (m0 + lm) * HD + 8];

    // Batch-load the 4 K-tile B fragments, then 4 WMMAs back-to-back.
    Frag kb[4];
    #pragma unroll
    for (int nt = 0; nt < 4; ++nt) {
        if (lh == 0) {   // K rows j=0..15; lanes>=16 (j=16..31) are zero padding
            const i32x4* kp = (const i32x4*)&K[base + (nt * 16 + lm) * HD];
            kb[nt].q[0] = kp[0]; kb[nt].q[1] = kp[1];
        } else {
            kb[nt].q[0] = zero; kb[nt].q[1] = zero;
        }
    }
    f32x8 s[4];
    #pragma unroll
    for (int nt = 0; nt < 4; ++nt) {
        f32x8 z = {0.f, 0.f, 0.f, 0.f, 0.f, 0.f, 0.f, 0.f};
        s[nt] = wmma_bf16(qa, kb[nt], z);
    }

    // Softmax over rows (each row lives in one lane-half at fixed r across 4 accs)
    u16t* pw = &pt[wave * 16 * PROW];
    #pragma unroll
    for (int r = 0; r < 8; ++r) {
        float v0 = s[0][r] * SCALE, v1 = s[1][r] * SCALE;
        float v2 = s[2][r] * SCALE, v3 = s[3][r] * SCALE;
        float mx = fmaxf(fmaxf(v0, v1), fmaxf(v2, v3));
        #pragma unroll
        for (int d = 1; d < 16; d <<= 1) mx = fmaxf(mx, __shfl_xor(mx, d, 32));
        v0 = __expf(v0 - mx); v1 = __expf(v1 - mx);
        v2 = __expf(v2 - mx); v3 = __expf(v3 - mx);
        float sm = v0 + v1 + v2 + v3;
        #pragma unroll
        for (int d = 1; d < 16; d <<= 1) sm += __shfl_xor(sm, d, 32);
        const float inv = 1.0f / sm;
        const int row = r + 8 * lh;
        pw[row * PROW +  0 + lm] = f2bf(v0 * inv);
        pw[row * PROW + 16 + lm] = f2bf(v1 * inv);
        pw[row * PROW + 32 + lm] = f2bf(v2 * inv);
        pw[row * PROW + 48 + lm] = f2bf(v3 * inv);
    }
    __syncthreads();

    // O = P (16x64) * V (64x16), two K=32 steps; batch loads then 2 WMMAs.
    const size_t vtb = (size_t)unit * (HD * TOK);
    Frag pa[2], vb[2];
    #pragma unroll
    for (int ks = 0; ks < 2; ++ks) {
        pa[ks].q[0] = *(const i32x4*)&pw[lm * PROW + ks * 32 + 16 * lh];
        pa[ks].q[1] = *(const i32x4*)&pw[lm * PROW + ks * 32 + 16 * lh + 8];
        const i32x4* vp = (const i32x4*)&Vt[vtb + lm * TOK + ks * 32 + 16 * lh];
        vb[ks].q[0] = vp[0]; vb[ks].q[1] = vp[1];
    }
    f32x8 o = {0.f, 0.f, 0.f, 0.f, 0.f, 0.f, 0.f, 0.f};
    #pragma unroll
    for (int ks = 0; ks < 2; ++ks) o = wmma_bf16(pa[ks], vb[ks], o);

    // Store O into token-major activation buffer (bf16)
    const size_t obase = ((size_t)b * SPT + (size_t)win * TOK) * DIMC + h * HD + lm;
    #pragma unroll
    for (int r = 0; r < 8; ++r)
        Ao[obase + (size_t)(m0 + r + 8 * lh) * DIMC] = f2bf(o[r]);
}

// ---------------------------------------------------------------------------
// Kernel 3: output projection. out[b][c][l] = Ao[b][l][:] . Wout[c][:] + bout[c]
// where l is the window-ordered token index reinterpreted as row-major spatial
// (faithful to the reference). One block per 64 tokens.
// ---------------------------------------------------------------------------
#define APAD 136
__global__ __launch_bounds__(256) void proj_kernel(
        const u16t* __restrict__ Ao, const u16t* __restrict__ wo,
        const float* __restrict__ bout, float* __restrict__ out) {
    __shared__ u16t at[TOK * APAD];
    const int tid = threadIdx.x, wave = tid >> 5, lane = tid & 31;
    const int lh = lane >> 4, lm = lane & 15;
    const size_t mbase = (size_t)blockIdx.x * 64;

    // Stage 64x128 bf16 activation tile (fully coalesced 16B chunks)
    for (int it = 0; it < 4; ++it) {
        int idx = it * 256 + tid;
        int flat = idx * 8;
        int r = flat >> 7, c = flat & 127;
        *(i32x4*)&at[r * APAD + c] = *(const i32x4*)&Ao[(mbase + r) * DIMC + c];
    }
    __syncthreads();

    const int n0 = wave * 16;
    Frag bfr[4];
    #pragma unroll
    for (int k = 0; k < 4; ++k) {   // cache B fragments, reused across 4 M-tiles
        const i32x4* bp = (const i32x4*)&wo[(n0 + lm) * DIMC + k * 32 + 16 * lh];
        bfr[k].q[0] = bp[0]; bfr[k].q[1] = bp[1];
    }
    const float bias = bout[n0 + lm];

    for (int mt = 0; mt < 4; ++mt) {
        // Batch the 8 ds_read_b128s, then 4 WMMAs back-to-back.
        Frag a[4];
        #pragma unroll
        for (int k = 0; k < 4; ++k) {
            a[k].q[0] = *(const i32x4*)&at[(mt * 16 + lm) * APAD + k * 32 + 16 * lh];
            a[k].q[1] = *(const i32x4*)&at[(mt * 16 + lm) * APAD + k * 32 + 16 * lh + 8];
        }
        f32x8 acc = {0.f, 0.f, 0.f, 0.f, 0.f, 0.f, 0.f, 0.f};
        #pragma unroll
        for (int k = 0; k < 4; ++k) acc = wmma_bf16(a[k], bfr[k], acc);

        // tokens mbase+mt*16+8*lh+r (contiguous in l), channel n0+lm (strided)
        const size_t tok0 = mbase + mt * 16 + 8 * lh;
        const int bb = (int)(tok0 / SPT);
        const size_t lsp = tok0 % SPT;
        float* dst = out + (size_t)(bb * DIMC + n0 + lm) * SPT + lsp;
        f32x4 lo = {acc[0] + bias, acc[1] + bias, acc[2] + bias, acc[3] + bias};
        f32x4 hi = {acc[4] + bias, acc[5] + bias, acc[6] + bias, acc[7] + bias};
        *(f32x4*)dst       = lo;
        *(f32x4*)(dst + 4) = hi;
    }
}

// ---------------------------------------------------------------------------
extern "C" void kernel_launch(void* const* d_in, const int* in_sizes, int n_in,
                              void* d_out, int out_size, void* d_ws, size_t ws_size,
                              hipStream_t stream) {
    (void)in_sizes; (void)n_in; (void)out_size; (void)ws_size;
    const float* x    = (const float*)d_in[0];
    const float* Wqkv = (const float*)d_in[1];
    const float* Wout = (const float*)d_in[2];
    const float* bout = (const float*)d_in[3];
    float* out = (float*)d_out;

    char* ws = (char*)d_ws;
    u16t* wq = (u16t*)ws;                         // 98304 B
    u16t* wo = (u16t*)(ws + 98304);               // 32768 B  (ends at 131072)
    const size_t qkvN = (size_t)NB * HEADS * NWIN * TOK * HD;  // 28,311,552 el
    u16t* Q  = (u16t*)(ws + 131072);
    u16t* K  = Q  + qkvN;
    u16t* Vt = K  + qkvN;
    u16t* Ao = Vt + qkvN;

    hipLaunchKernelGGL(cvt_weights, dim3(192), dim3(256), 0, stream, Wqkv, Wout, wq, wo);
    hipLaunchKernelGGL(qkv_kernel,  dim3(NB * NWIN), dim3(256), 0, stream, x, wq, Q, K, Vt);
    hipLaunchKernelGGL(attn_kernel, dim3(NB * HEADS * NWIN), dim3(128), 0, stream, Q, K, Vt, Ao);
    hipLaunchKernelGGL(proj_kernel, dim3((NB * SPT) / 64), dim3(256), 0, stream, Ao, wo, bout, out);
}
